// LanguageModel_34454227648823
// MI455X (gfx1250) — matrix operand order repaired and checked by reading, not compiled, not run on hardware
//
#include <hip/hip_runtime.h>
#include <hip/hip_bf16.h>

#define VOCAB   50257
#define DIM     512
#define BATCH   64
#define SEQ     256
#define RNN_WGS 32        // each WG owns 16 hidden columns (32*16 = 512)
#define RNN_THR 128       // 4 waves: one 16-row batch tile each (4*16 = 64)
#define NTILES  3142      // ceil(50257/16)

typedef __attribute__((ext_vector_type(2)))  float    v2f;
typedef __attribute__((ext_vector_type(8)))  float    v8f;
typedef __attribute__((ext_vector_type(16))) _Float16 v16h;

// ---------------------------------------------------------------------------
// branch-free tanh on the serial recurrence path:
//   tanh(x) = copysign(1 - 2/(exp(2|x|)+1), x)
// v_exp_f32 + v_rcp_f32, no exec-mask divergence, saturates to +-1, inf-safe.
// ---------------------------------------------------------------------------
__device__ __forceinline__ float fast_tanh(float x) {
  float ax = __builtin_fabsf(x);
  float e  = __expf(2.0f * ax);                       // v_exp_f32 path
  float t  = 1.0f - 2.0f * __builtin_amdgcn_rcpf(e + 1.0f);
  return __builtin_copysignf(t, x);
}

// ---------------------------------------------------------------------------
// setup: zero the h-state ping-pong buffers (h0 = 0) and barrier control words
// ---------------------------------------------------------------------------
__global__ void setup_kernel(float* __restrict__ hbufs, int n, int* __restrict__ ctrl) {
  int i = blockIdx.x * blockDim.x + threadIdx.x;
  if (i < n) hbufs[i] = 0.0f;
  if (i == 0) { ctrl[0] = 0; ctrl[1] = 0; }
}

// ---------------------------------------------------------------------------
// device-wide sense-reversing barrier (32 resident WGs)
// ---------------------------------------------------------------------------
__device__ __forceinline__ void grid_barrier(int* __restrict__ ctrl, int nwg) {
  __threadfence();               // publish h-slice stores (device scope)
  __syncthreads();
  if (threadIdx.x == 0) {
    int* counter = ctrl;
    int* gen     = ctrl + 1;
    int  g = __hip_atomic_load(gen, __ATOMIC_RELAXED, __HIP_MEMORY_SCOPE_AGENT);
    int  a = __hip_atomic_fetch_add(counter, 1, __ATOMIC_ACQ_REL, __HIP_MEMORY_SCOPE_AGENT);
    if (a == nwg - 1) {
      __hip_atomic_store(counter, 0, __ATOMIC_RELAXED, __HIP_MEMORY_SCOPE_AGENT);
      __hip_atomic_fetch_add(gen, 1, __ATOMIC_RELEASE, __HIP_MEMORY_SCOPE_AGENT);
    } else {
      while (__hip_atomic_load(gen, __ATOMIC_ACQUIRE, __HIP_MEMORY_SCOPE_AGENT) == g)
        __builtin_amdgcn_s_sleep(2);
    }
  }
  __syncthreads();               // acquire-inv hit WGP$, shared by all block waves
}

// ---------------------------------------------------------------------------
// RNN recurrence: 32 WGs, weights LDS-resident for all 256 steps, f32 WMMA.
// WG b computes hidden columns [16b, 16b+16); wave w computes batch rows
// [16w, 16w+16). Per layer: one 16x16 f32 accumulator over K=512 (input path)
// + K=512 (recurrent path), tanh, store slice, grid barrier.
// ---------------------------------------------------------------------------
__global__ void __launch_bounds__(RNN_THR)
rnn_kernel(const int*   __restrict__ x,    const float* __restrict__ emb,
           const float* __restrict__ Wih0, const float* __restrict__ Whh0,
           const float* __restrict__ bih0, const float* __restrict__ bhh0,
           const float* __restrict__ Wih1, const float* __restrict__ Whh1,
           const float* __restrict__ bih1, const float* __restrict__ bhh1,
           float* __restrict__ h1buf, float* __restrict__ h2buf,
           int* __restrict__ ctrl)
{
  // W^T slices, pair-interleaved: idx(k,n) = (k>>1)*32 + 2n + (k&1)
  // -> each lane's B operand (k,k+1 at column n) is one conflict-free b64.
  __shared__ float wlds[4][16 * DIM];           // 128 KB of 320 KB LDS
  const int tid = threadIdx.x;
  const int n0  = blockIdx.x * 16;
  {
    const float* Ws[4] = {Wih0, Whh0, Wih1, Whh1};
    for (int m = 0; m < 4; ++m)
      for (int i = tid; i < 16 * DIM; i += RNN_THR) {
        int n = i >> 9, k = i & (DIM - 1);
        wlds[m][(k >> 1) * 32 + n * 2 + (k & 1)] = Ws[m][(size_t)(n0 + n) * DIM + k];
      }
  }
  __syncthreads();

  const int wid  = tid >> 5;
  const int lane = tid & 31;
  const int ln   = lane & 15;
  const int hi   = lane >> 4;      // hi half-wave carries K+2 (A and B)
  const int koff = hi * 2;
  const int m0   = wid * 16;
  const int row  = m0 + ln;        // batch row this lane supplies A data for

  const float bias0 = bih0[n0 + ln] + bhh0[n0 + ln];
  const float bias1 = bih1[n0 + ln] + bhh1[n0 + ln];

  for (int t = 0; t < SEQ; ++t) {
    const float* h1p = h1buf + (size_t)( t      & 1) * (BATCH * DIM);
    float*       h1n = h1buf + (size_t)((t + 1) & 1) * (BATCH * DIM);
    const float* h2p = h2buf + (size_t)( t      & 1) * (BATCH * DIM);
    float*       h2n = h2buf + (size_t)((t + 1) & 1) * (BATCH * DIM);

    // ---- layer 0: h1 = tanh(x_t W_ih0^T + h1_prev W_hh0^T + b) ----
    v8f acc;
    #pragma unroll
    for (int v = 0; v < 8; ++v) acc[v] = bias0;   // N = ln for every C VGPR
    const float* xrow = emb + (size_t)x[row * SEQ + t] * DIM;  // fused gather
    const float* hrow = h1p + (size_t)row * DIM;
    #pragma unroll 4
    for (int k0 = 0; k0 < DIM; k0 += 4) {
      const int bidx = ((k0 >> 1) + hi) * 32 + ln * 2;
      v2f aX = *(const v2f*)(xrow + k0 + koff);
      v2f b0 = *(const v2f*)&wlds[0][bidx];
      acc = __builtin_amdgcn_wmma_f32_16x16x4_f32(false, aX, false, b0,
                                                  (short)0, acc, false, false);
      v2f aH = *(const v2f*)(hrow + k0 + koff);
      v2f b1 = *(const v2f*)&wlds[1][bidx];
      acc = __builtin_amdgcn_wmma_f32_16x16x4_f32(false, aH, false, b1,
                                                  (short)0, acc, false, false);
    }
    #pragma unroll
    for (int v = 0; v < 8; ++v)
      h1n[(size_t)(m0 + v + hi * 8) * DIM + n0 + ln] = fast_tanh(acc[v]);

    grid_barrier(ctrl, RNN_WGS);   // all WGs see full h1_t

    // ---- layer 1: h2 = tanh(h1_t W_ih1^T + h2_prev W_hh1^T + b) ----
    #pragma unroll
    for (int v = 0; v < 8; ++v) acc[v] = bias1;
    const float* arow = h1n + (size_t)row * DIM;
    const float* brow = h2p + (size_t)row * DIM;
    #pragma unroll 4
    for (int k0 = 0; k0 < DIM; k0 += 4) {
      const int bidx = ((k0 >> 1) + hi) * 32 + ln * 2;
      v2f aX = *(const v2f*)(arow + k0 + koff);
      v2f b2 = *(const v2f*)&wlds[2][bidx];
      acc = __builtin_amdgcn_wmma_f32_16x16x4_f32(false, aX, false, b2,
                                                  (short)0, acc, false, false);
      v2f aH = *(const v2f*)(brow + k0 + koff);
      v2f b3 = *(const v2f*)&wlds[3][bidx];
      acc = __builtin_amdgcn_wmma_f32_16x16x4_f32(false, aH, false, b3,
                                                  (short)0, acc, false, false);
    }
    #pragma unroll
    for (int v = 0; v < 8; ++v)
      h2n[(size_t)(m0 + v + hi * 8) * DIM + n0 + ln] = fast_tanh(acc[v]);

    grid_barrier(ctrl, RNN_WGS);   // all WGs see full h2_t
  }
}

// ---------------------------------------------------------------------------
// Classifier: out = h2 @ W_cls^T + b_cls  (64 x 50257, K = 512).
// HBM-roofline bound on the single streaming read of W_cls (103 MB f32
// ~ 4.4 us @ 23.3 TB/s): convert f32 -> f16 in registers, f32-accumulate
// v_wmma_f32_16x16x32_f16. One 16x16 tile per wave, 16 WMMAs per tile.
// ---------------------------------------------------------------------------
__global__ void __launch_bounds__(256)
cls_kernel(const float* __restrict__ h2, const float* __restrict__ Wc,
           const float* __restrict__ bc, float* __restrict__ out)
{
  const int gw    = blockIdx.x * 8 + (threadIdx.x >> 5);
  const int mtile = gw & 3;
  const int ntile = gw >> 2;              // 0 .. 3141
  const int lane  = threadIdx.x & 31;
  const int ln    = lane & 15;
  const int hi    = lane >> 4;
  const int m0    = mtile * 16;
  const int n0    = ntile * 16;
  const int ncol  = n0 + ln;
  const int nld   = ncol < VOCAB ? ncol : (VOCAB - 1);   // clamp tail loads

  v8f acc;
  const float bias = bc[nld];
  #pragma unroll
  for (int v = 0; v < 8; ++v) acc[v] = bias;

  const float* arow = h2 + (size_t)(m0 + ln) * DIM;      // A row (h2)
  const float* brow = Wc + (size_t)nld * DIM;            // B col n = W_cls row n

  for (int k0 = 0; k0 < DIM; k0 += 32) {
    v16h A, B;
    #pragma unroll
    for (int j = 0; j < 8; ++j) {
      // A 16x32 f16, this code's contiguous k order: vgpr j -> K=2j (lanes 0-15) / K=16+2j (lanes 16-31)
      const int ka = k0 + 2 * j + hi * 16;
      float2 av = *(const float2*)(arow + ka);
      A[2 * j]     = (_Float16)av.x;
      A[2 * j + 1] = (_Float16)av.y;
      // B 32x16 f16, same contiguous k order as A: vgpr j -> K=2j (lanes 0-15) / K=16+2j (lanes 16-31)
      const int kb = k0 + 2 * j + hi * 16;
      float2 bv = *(const float2*)(brow + kb);
      B[2 * j]     = (_Float16)bv.x;
      B[2 * j + 1] = (_Float16)bv.y;
    }
    acc = __builtin_amdgcn_wmma_f32_16x16x32_f16(false, A, false, B,
                                                 (short)0, acc, false, false);
  }

  #pragma unroll
  for (int v = 0; v < 8; ++v) {
    const int r = m0 + v + hi * 8;
    if (ncol < VOCAB) out[(size_t)r * VOCAB + ncol] = acc[v];
  }
}

// ---------------------------------------------------------------------------
extern "C" void kernel_launch(void* const* d_in, const int* in_sizes, int n_in,
                              void* d_out, int out_size, void* d_ws, size_t ws_size,
                              hipStream_t stream) {
  const int*   x    = (const int*)  d_in[0];
  const float* emb  = (const float*)d_in[1];
  const float* Wih0 = (const float*)d_in[2];
  const float* Whh0 = (const float*)d_in[3];
  const float* bih0 = (const float*)d_in[4];
  const float* bhh0 = (const float*)d_in[5];
  const float* Wih1 = (const float*)d_in[6];
  const float* Whh1 = (const float*)d_in[7];
  const float* bih1 = (const float*)d_in[8];
  const float* bhh1 = (const float*)d_in[9];
  const float* Wcls = (const float*)d_in[10];
  const float* bcls = (const float*)d_in[11];
  float* out = (float*)d_out;

  // workspace: [0,256) barrier ctrl, then h1[2][64][512], h2[2][64][512]
  int*   ctrl = (int*)d_ws;
  float* h1   = (float*)((char*)d_ws + 256);
  float* h2   = h1 + 2 * BATCH * DIM;

  const int nz = 4 * BATCH * DIM;
  setup_kernel<<<(nz + 255) / 256, 256, 0, stream>>>(h1, nz, ctrl);

  rnn_kernel<<<RNN_WGS, RNN_THR, 0, stream>>>(x, emb, Wih0, Whh0, bih0, bhh0,
                                              Wih1, Whh1, bih1, bhh1,
                                              h1, h2, ctrl);

  // final h2 lives in ping-pong buffer (SEQ & 1) == 0
  cls_kernel<<<(NTILES * 4) / 8, 256, 0, stream>>>(h2, Wcls, bcls, out);
}
